// GlobalAttention_71244917506827
// MI455X (gfx1250) — matrix operand order repaired and checked by reading, not compiled, not run on hardware
//
#include <hip/hip_runtime.h>

// ---------------------------------------------------------------------------
// Linear attention for MI455X (gfx1250, wave32, WMMA bf16 16x16x32,
// async global->LDS DMA with double buffering)
// ---------------------------------------------------------------------------
typedef __bf16 bf16;
typedef __attribute__((ext_vector_type(16))) bf16  v16bf;
typedef __attribute__((ext_vector_type(8)))  bf16  v8bf;
typedef __attribute__((ext_vector_type(8)))  float v8f;

#define DIM 256
#define NH 8
#define HD 32
#define CHUNK 32            // rows per WMMA macro-step
#define CPW 8               // chunks per workgroup in pass A
#define ROWS_PER_WG (CHUNK * CPW)
#define EPS 1e-8f

__device__ __forceinline__ float sigmoidf_(float x) {
    return 1.0f / (1.0f + __expf(-x));
}

// A-fragment (16x32 bf16): lane r=lane%16 is the M row, halves split K.
// elements 0..7  = K[8*h2 .. 8*h2+7], elements 8..15 = K[16+8*h2 .. 23+8*h2]
__device__ __forceinline__ v16bf load_afrag(const bf16* __restrict__ src,
                                            int ld, int row, int h2, int k0) {
    const bf16* p = src + (size_t)row * ld + k0 + 8 * h2;
    v8bf lo = *(const v8bf*)p;
    v8bf hi = *(const v8bf*)(p + 16);
    return __builtin_shufflevector(lo, hi, 0,1,2,3,4,5,6,7,8,9,10,11,12,13,14,15);
}

// B-fragment (32x16 bf16): lane c=lane%16 is the N column; source is laid out
// with one "row" per output column (i.e. W row-major for x@W^T, kv[h][m][d]).
// elements 0..7 = K[8*h2 .. 8*h2+7], elements 8..15 = K[16+8*h2 .. 23+8*h2]
__device__ __forceinline__ v16bf load_bfrag(const bf16* __restrict__ src,
                                            int ld, int col, int h2, int k0) {
    const bf16* p = src + (size_t)col * ld + k0 + 8 * h2;
    v8bf lo = *(const v8bf*)p;
    v8bf hi = *(const v8bf*)(p + 16);
    return __builtin_shufflevector(lo, hi, 0,1,2,3,4,5,6,7,8,9,10,11,12,13,14,15);
}

__device__ __forceinline__ v8f wmma_bf16(v16bf a, v16bf b, v8f c) {
    return __builtin_amdgcn_wmma_f32_16x16x32_bf16(false, a, false, b,
                                                   (short)0, c, false, false);
}

// Async DMA of one 32x256 bf16 chunk (16 KB) into LDS: 64B per thread as
// 4 x GLOBAL_LOAD_ASYNC_TO_LDS_B128 (tracked by ASYNCcnt, no VGPR round-trip).
// Flat LDS addresses: low 32 bits are the LDS byte offset (ISA 10.2 aperture).
__device__ __forceinline__ void stage_chunk_async(bf16* lds_dst,
                                                  const bf16* g_src) {
    unsigned int l = (unsigned int)(size_t)lds_dst + threadIdx.x * 64u;
    const char*  g = (const char*)g_src + threadIdx.x * 64u;
    #pragma unroll
    for (int i = 0; i < 4; ++i)
        asm volatile("global_load_async_to_lds_b128 %0, %1, off"
                     :: "v"(l + i * 16u), "v"(g + i * 16) : "memory");
}

__device__ __forceinline__ void wait_async_all() {
    asm volatile("s_wait_asynccnt 0x0" ::: "memory");
}

// ---------------------------------------------------------------------------
// K0/K1: fp32 -> bf16 conversion (x and the three weight matrices)
// ---------------------------------------------------------------------------
__global__ __launch_bounds__(256) void cvt_bf16_kernel(const float* __restrict__ src,
                                                       bf16* __restrict__ dst, int n) {
    int i = (blockIdx.x * blockDim.x + threadIdx.x) * 4;
    if (i + 3 < n) {
        float4 f = *(const float4*)(src + i);
        dst[i + 0] = (bf16)f.x;
        dst[i + 1] = (bf16)f.y;
        dst[i + 2] = (bf16)f.z;
        dst[i + 3] = (bf16)f.w;
    } else {
        for (int j = i; j < n; ++j) dst[j] = (bf16)src[j];
    }
}

// ---------------------------------------------------------------------------
// K2: pass A. Per WG: 256 rows. wave w == head w (32 cols).
//   k = sigmoid(x Wk^T + bk), v = x Wv^T + bv  via WMMA (K=256)
//   kv[h] += k^T v (32x32, WMMA K=32), ksum[h] += sum_n k
// Async double-buffered x staging; deterministic per-block partials.
// ---------------------------------------------------------------------------
__global__ __launch_bounds__(256) void kv_pass_kernel(
    const bf16* __restrict__ xb,
    const bf16* __restrict__ wkb, const bf16* __restrict__ wvb,
    const float* __restrict__ bk, const float* __restrict__ bv,
    float* __restrict__ kv_part, float* __restrict__ ksum_part, int n) {

    __shared__ bf16 lds_x[2][CHUNK][DIM];     // 32 KB double-buffered x
    __shared__ bf16 lds_kT[NH][HD][CHUNK];    // 16 KB per-wave k^T (m,n)
    __shared__ bf16 lds_vT[NH][HD][CHUNK];    // 16 KB per-wave v^T (d,n)

    const int w    = threadIdx.x >> 5;        // wave id == head id
    const int lane = threadIdx.x & 31;
    const int c16  = lane & 15;
    const int h2   = lane >> 4;

    v8f   kvacc[2][2] = {};                   // kv[h] 32x32 accumulator
    float ksacc[2]    = {0.f, 0.f};

    float bkc[2], bvc[2];
    for (int ct = 0; ct < 2; ++ct) {
        int j = w * HD + ct * 16 + c16;
        bkc[ct] = bk[j];
        bvc[ct] = bv[j];
    }

    const int base = blockIdx.x * ROWS_PER_WG;
    const int nch  = min(CPW, (n - base + CHUNK - 1) / CHUNK);  // uniform

    // prologue: DMA chunk 0 into buffer 0
    stage_chunk_async(&lds_x[0][0][0], xb + (size_t)base * DIM);
    wait_async_all();
    __syncthreads();

    for (int ch = 0; ch < nch; ++ch) {
        const int buf = ch & 1;
        // prefetch next chunk into the other buffer while we compute
        if (ch + 1 < nch)
            stage_chunk_async(&lds_x[buf ^ 1][0][0],
                              xb + (size_t)(base + (ch + 1) * CHUNK) * DIM);

        // ---- GEMM: k and v tiles (2 row-tiles x 2 col-tiles, K=256) ----
        v8f ka[2][2] = {}, va[2][2] = {};
        #pragma unroll
        for (int kt = 0; kt < 8; ++kt) {
            const int k0 = kt * 32;
            v16bf afr[2];
            #pragma unroll
            for (int rt = 0; rt < 2; ++rt)
                afr[rt] = load_afrag(&lds_x[buf][0][0], DIM, rt * 16 + c16, h2, k0);
            #pragma unroll
            for (int ct = 0; ct < 2; ++ct) {
                const int j = w * HD + ct * 16 + c16;
                v16bf bkf = load_bfrag(wkb, DIM, j, h2, k0);
                v16bf bvf = load_bfrag(wvb, DIM, j, h2, k0);
                #pragma unroll
                for (int rt = 0; rt < 2; ++rt) {
                    ka[rt][ct] = wmma_bf16(afr[rt], bkf, ka[rt][ct]);
                    va[rt][ct] = wmma_bf16(afr[rt], bvf, va[rt][ct]);
                }
            }
        }

        // ---- epilogue: bias + sigmoid, ksum, transpose-scatter to LDS ----
        #pragma unroll
        for (int ct = 0; ct < 2; ++ct) {
            #pragma unroll
            for (int rt = 0; rt < 2; ++rt) {
                float s = 0.f;
                #pragma unroll
                for (int e = 0; e < 8; ++e) {
                    const int nn = rt * 16 + h2 * 8 + e;      // row within chunk
                    const int cc = ct * 16 + c16;             // col within head
                    float kvl = sigmoidf_(ka[rt][ct][e] + bkc[ct]);
                    float vvl = va[rt][ct][e] + bvc[ct];
                    s += kvl;
                    lds_kT[w][cc][nn] = (bf16)kvl;
                    lds_vT[w][cc][nn] = (bf16)vvl;
                }
                ksacc[ct] += s;
            }
        }
        asm volatile("s_wait_dscnt 0" ::: "memory");  // same-wave LDS RAW

        // ---- kv[h] += k^T v : A = k^T (m,n), B = v (n,d) via v^T rows ----
        #pragma unroll
        for (int mt = 0; mt < 2; ++mt) {
            v16bf afr = load_afrag(&lds_kT[w][0][0], CHUNK, mt * 16 + c16, h2, 0);
            #pragma unroll
            for (int dt = 0; dt < 2; ++dt) {
                v16bf bfr = load_bfrag(&lds_vT[w][0][0], CHUNK, dt * 16 + c16, h2, 0);
                kvacc[mt][dt] = wmma_bf16(afr, bfr, kvacc[mt][dt]);
            }
        }

        // next buffer ready + all waves done with current buffers
        wait_async_all();
        __syncthreads();
    }

    // ---- write deterministic per-block partials ----
    float* kvp = kv_part + (size_t)blockIdx.x * (NH * HD * HD);
    #pragma unroll
    for (int mt = 0; mt < 2; ++mt)
        #pragma unroll
        for (int dt = 0; dt < 2; ++dt)
            #pragma unroll
            for (int e = 0; e < 8; ++e) {
                const int m = mt * 16 + h2 * 8 + e;
                const int d = dt * 16 + c16;
                kvp[w * HD * HD + m * HD + d] = kvacc[mt][dt][e];
            }
    #pragma unroll
    for (int ct = 0; ct < 2; ++ct) {
        float tot = ksacc[ct] + __shfl_xor(ksacc[ct], 16);
        if (h2 == 0)
            ksum_part[(size_t)blockIdx.x * DIM + w * HD + ct * 16 + c16] = tot;
    }
}

// ---------------------------------------------------------------------------
// K3: reduce block partials -> kv (bf16, [h][m][d] for direct B-fragments)
//     and ksum (f32)
// ---------------------------------------------------------------------------
__global__ __launch_bounds__(256) void reduce_kernel(
    const float* __restrict__ kv_part, const float* __restrict__ ksum_part,
    bf16* __restrict__ kvf, float* __restrict__ ksf, int nblk) {
    int t = blockIdx.x * blockDim.x + threadIdx.x;
    if (t < NH * HD * HD) {
        float s = 0.f;
        for (int b = 0; b < nblk; ++b) s += kv_part[(size_t)b * (NH * HD * HD) + t];
        kvf[t] = (bf16)s;
    } else if (t < NH * HD * HD + DIM) {
        int i = t - NH * HD * HD;
        float s = 0.f;
        for (int b = 0; b < nblk; ++b) s += ksum_part[(size_t)b * DIM + i];
        ksf[i] = s;
    }
}

// ---------------------------------------------------------------------------
// K4: pass B. Per WG: 32 rows, wave w == head w.
//   q = sigmoid(x Wq^T + bq)  (WMMA, K=256; recomputed, no q spill)
//   num = q kv^T (WMMA, K=32), den = q . ksum, out = num / (den + eps)
// ---------------------------------------------------------------------------
__global__ __launch_bounds__(256) void out_pass_kernel(
    const bf16* __restrict__ xb,
    const bf16* __restrict__ wqb, const float* __restrict__ bq,
    const bf16* __restrict__ kvf, const float* __restrict__ ksf,
    float* __restrict__ out, int n) {

    __shared__ bf16  lds_x[CHUNK][DIM];       // 16 KB
    __shared__ bf16  lds_q[NH][CHUNK][HD];    // 16 KB per-wave q tile [row][d]
    __shared__ float lds_den[NH][CHUNK];
    __shared__ float lds_ks[NH * HD];

    const int w    = threadIdx.x >> 5;
    const int lane = threadIdx.x & 31;
    const int c16  = lane & 15;
    const int h2   = lane >> 4;
    const int n0   = blockIdx.x * CHUNK;

    lds_ks[threadIdx.x] = ksf[threadIdx.x];   // 256 values, one per thread
    stage_chunk_async(&lds_x[0][0], xb + (size_t)n0 * DIM);
    wait_async_all();
    __syncthreads();

    // ---- q GEMM (K=256) ----
    v8f qa[2][2] = {};
    #pragma unroll
    for (int kt = 0; kt < 8; ++kt) {
        const int k0 = kt * 32;
        v16bf afr[2];
        #pragma unroll
        for (int rt = 0; rt < 2; ++rt)
            afr[rt] = load_afrag(&lds_x[0][0], DIM, rt * 16 + c16, h2, k0);
        #pragma unroll
        for (int ct = 0; ct < 2; ++ct) {
            const int j = w * HD + ct * 16 + c16;
            v16bf bqf = load_bfrag(wqb, DIM, j, h2, k0);
            #pragma unroll
            for (int rt = 0; rt < 2; ++rt)
                qa[rt][ct] = wmma_bf16(afr[rt], bqf, qa[rt][ct]);
        }
    }

    // ---- sigmoid epilogue, scatter q[row][d] to LDS ----
    #pragma unroll
    for (int ct = 0; ct < 2; ++ct) {
        const float bqc = bq[w * HD + ct * 16 + c16];
        #pragma unroll
        for (int rt = 0; rt < 2; ++rt)
            #pragma unroll
            for (int e = 0; e < 8; ++e) {
                const int row = rt * 16 + h2 * 8 + e;
                lds_q[w][row][ct * 16 + c16] = (bf16)sigmoidf_(qa[rt][ct][e] + bqc);
            }
    }
    asm volatile("s_wait_dscnt 0" ::: "memory");

    // ---- denominator: lane l handles row l of this wave's head ----
    {
        float den = 0.f;
        #pragma unroll
        for (int d = 0; d < HD; ++d)
            den += (float)lds_q[w][lane][d] * lds_ks[w * HD + d];
        lds_den[w][lane] = den;
    }

    // ---- numerator: num = q (16x32) x kv^T (32x16) per (rt, mt) tile ----
    v8f num[2][2] = {};
    #pragma unroll
    for (int rt = 0; rt < 2; ++rt) {
        v16bf afr = load_afrag(&lds_q[w][0][0], HD, rt * 16 + c16, h2, 0);
        #pragma unroll
        for (int mt = 0; mt < 2; ++mt) {
            v16bf bfr = load_bfrag(kvf + (size_t)w * HD * HD, HD, mt * 16 + c16, h2, 0);
            num[rt][mt] = wmma_bf16(afr, bfr, num[rt][mt]);
        }
    }
    asm volatile("s_wait_dscnt 0" ::: "memory");

    // ---- out = num / (den + eps) ----
    #pragma unroll
    for (int rt = 0; rt < 2; ++rt)
        #pragma unroll
        for (int mt = 0; mt < 2; ++mt)
            #pragma unroll
            for (int e = 0; e < 8; ++e) {
                const int row = rt * 16 + h2 * 8 + e;
                const int m   = mt * 16 + c16;
                out[(size_t)(n0 + row) * DIM + w * HD + m] =
                    num[rt][mt][e] / (lds_den[w][row] + EPS);
            }
}

// ---------------------------------------------------------------------------
extern "C" void kernel_launch(void* const* d_in, const int* in_sizes, int n_in,
                              void* d_out, int out_size, void* d_ws, size_t ws_size,
                              hipStream_t stream) {
    const float* x  = (const float*)d_in[0];
    const float* Wq = (const float*)d_in[1];
    const float* bq = (const float*)d_in[2];
    const float* Wk = (const float*)d_in[3];
    const float* bk = (const float*)d_in[4];
    const float* Wv = (const float*)d_in[5];
    const float* bv = (const float*)d_in[6];
    float* out = (float*)d_out;

    const int n    = in_sizes[0] / DIM;       // 100000 (multiple of 32)
    const int nblk = (n + ROWS_PER_WG - 1) / ROWS_PER_WG;

    char* ws = (char*)d_ws;
    size_t off = 0;
    auto alloc = [&](size_t bytes) {
        void* p = ws + off;
        off = (off + bytes + 255) & ~(size_t)255;
        return p;
    };
    bf16*  xb  = (bf16*)alloc((size_t)n * DIM * sizeof(bf16));
    bf16*  wqb = (bf16*)alloc((size_t)DIM * DIM * sizeof(bf16));
    bf16*  wkb = (bf16*)alloc((size_t)DIM * DIM * sizeof(bf16));
    bf16*  wvb = (bf16*)alloc((size_t)DIM * DIM * sizeof(bf16));
    float* kvp = (float*)alloc((size_t)nblk * NH * HD * HD * sizeof(float));
    float* ksp = (float*)alloc((size_t)nblk * DIM * sizeof(float));
    bf16*  kvf = (bf16*)alloc((size_t)NH * HD * HD * sizeof(bf16));
    float* ksf = (float*)alloc((size_t)DIM * sizeof(float));

    // K0/K1: fp32 -> bf16
    {
        int tot = n * DIM;
        cvt_bf16_kernel<<<(tot / 4 + 255) / 256, 256, 0, stream>>>(x, xb, tot);
        int wt = DIM * DIM;
        int wg = (wt / 4 + 255) / 256;
        cvt_bf16_kernel<<<wg, 256, 0, stream>>>(Wq, wqb, wt);
        cvt_bf16_kernel<<<wg, 256, 0, stream>>>(Wk, wkb, wt);
        cvt_bf16_kernel<<<wg, 256, 0, stream>>>(Wv, wvb, wt);
    }
    // K2: k/v GEMM + per-block kv, ksum partials
    kv_pass_kernel<<<nblk, 256, 0, stream>>>(xb, wkb, wvb, bk, bv, kvp, ksp, n);
    // K3: reduce partials
    reduce_kernel<<<(NH * HD * HD + DIM + 255) / 256, 256, 0, stream>>>(kvp, ksp, kvf, ksf, nblk);
    // K4: q GEMM + attention apply
    out_pass_kernel<<<n / CHUNK, 256, 0, stream>>>(xb, wqb, bq, kvf, ksf, out, n);
}
